// Seq2Seq_85126251806961
// MI455X (gfx1250) — compile-verified, matrix operand order checked by reading, not run on hardware
//
#include <hip/hip_runtime.h>
#include <hip/hip_bf16.h>

// ---------------------------------------------------------------------------
// Seq2Seq RNN (encoder/decoder, sigmoid) for MI455X (gfx1250, wave32, WMMA).
// Batch rows are independent through the recurrence, so each workgroup owns a
// 16-row batch tile and runs all T encoder + T decoder steps locally, holding
// double-buffered h / x / y tiles in LDS and driving v_wmma_f32_16x16x32_bf16.
// Weights are pre-packed to bf16 B-fragment order; the compiler keeps the
// Wxh slice register-resident (s_set_vgpr_msb / >512 VGPRs) and streams Whh
// from L2 each step.
// ---------------------------------------------------------------------------

#define S2S_N 4096
#define S2S_T 128
#define S2S_D 256
#define S2S_H 512
#define S2S_O 256

typedef __attribute__((ext_vector_type(16))) __bf16        v16bf;
typedef __attribute__((ext_vector_type(8)))  float         v8f;
typedef __attribute__((ext_vector_type(4)))  unsigned int  u32x4;

union BfFrag { v16bf v; u32x4 q[2]; };

// RNE f32->bf16 (used once, for weight packing)
__device__ __forceinline__ unsigned short f2bf_rne(float f) {
    unsigned int u = __float_as_uint(f);
    unsigned int r = u + 0x7fffu + ((u >> 16) & 1u);
    return (unsigned short)(r >> 16);
}
// truncating f32->bf16 (hot path)
__device__ __forceinline__ unsigned short f2bf_tr(float f) {
    return (unsigned short)(__float_as_uint(f) >> 16);
}

// A fragment (16x32 bf16): lane holds two 8-element K chunks 16 elements apart.
__device__ __forceinline__ v16bf load_a_frag(const unsigned short* p) {
    BfFrag f;
    f.q[0] = *(const u32x4*)(p);        // K chunk 0 (16B)
    f.q[1] = *(const u32x4*)(p + 16);   // K chunk 1 (16B), +16 elements
    return f.v;
}

// B fragment, pre-swizzled in global memory: lane reads 32 contiguous bytes.
__device__ __forceinline__ v16bf load_b_frag(const unsigned short* p) {
    BfFrag f;
    f.q[0] = *(const u32x4*)(p);
    f.q[1] = *(const u32x4*)(p + 8);
    return f.v;
}

#define WMMA_BF16(a, b, c) \
    __builtin_amdgcn_wmma_f32_16x16x32_bf16(false, (a), false, (b), (short)0, (c), false, false)

// ---------------------------------------------------------------------------
// Pack a KxNn fp32 weight matrix (row-major, W[k][n]) into bf16 B-fragment
// order: tiles (ntile,ktile), 512 elements each; within a tile, lane L holds
// column n = ntile*16 + (L&15) with K chunks {k0 + (L>>4)*8 .. +8} and
// {k0 + 16 + (L>>4)*8 .. +8}, stored contiguously (16 bf16 per lane).
// ---------------------------------------------------------------------------
__global__ void pack_b_kernel(const float* __restrict__ src,
                              unsigned short* __restrict__ dst,
                              int K, int Nn) {
    int e = blockIdx.x * 256 + threadIdx.x;
    int total = K * Nn;
    if (e >= total) return;
    int ktiles = K >> 5;
    int tile  = e >> 9;        // /512
    int rem   = e & 511;
    int laneL = rem >> 4;
    int j     = rem & 15;
    int ntile = tile / ktiles;
    int ktile = tile % ktiles;
    int n     = ntile * 16 + (laneL & 15);
    int half  = laneL >> 4;
    int kin   = (j < 8) ? (half * 8 + j) : (16 + half * 8 + (j - 8));
    int k     = ktile * 32 + kin;
    dst[e] = f2bf_rne(src[(size_t)k * Nn + n]);
}

// ---------------------------------------------------------------------------
// Main kernel: one 16-row batch tile per workgroup, 4 wave32s.
// ---------------------------------------------------------------------------
__launch_bounds__(128, 1)
__global__ void seq2seq_main_kernel(const float* __restrict__ x,      // (N,T,D) fp32
                                    const float* __restrict__ y0,     // (1,D)   fp32
                                    const unsigned short* __restrict__ eWxhP, // packed bf16
                                    const unsigned short* __restrict__ eWhhP,
                                    const unsigned short* __restrict__ dWxhP,
                                    const unsigned short* __restrict__ dWhhP,
                                    const unsigned short* __restrict__ dWhoP,
                                    float* __restrict__ out) {        // (T,N,O) fp32
    // Double-buffered, padded LDS tiles (+8 shorts of stride padding keeps
    // ds_load_b128 bank-conflict-free across the 16 rows).
    __shared__ __align__(16) unsigned short hbuf[2][16][S2S_H + 8]; // h / s (bf16)
    __shared__ __align__(16) unsigned short iob[2][16][S2S_D + 8];  // x_t / y_t (bf16)

    const int tid  = threadIdx.x;
    const int lane = tid & 31;
    const int wave = tid >> 5;
    const int half = lane >> 4;      // K-half selector for A fragments
    const int nl   = lane & 15;      // row (A) / column (B,C) within tile
    const int n0   = blockIdx.x * 16;

    const int xrow = tid >> 3;            // 0..15 : staging row
    const int xseg = (tid & 7) * 32;      // 32 floats per thread

    const v8f vzero = {0.f, 0.f, 0.f, 0.f, 0.f, 0.f, 0.f, 0.f};

    // h[parity 0] = 0
    for (int i = tid; i < 16 * (S2S_H + 8); i += 128) (&hbuf[0][0][0])[i] = 0;

    // stage x_{t=0} into iob[0]
    {
        const float* src = x + ((size_t)(n0 + xrow) * S2S_T + 0) * S2S_D + xseg;
#pragma unroll
        for (int c = 0; c < 32; c += 4) {
            float4 f = *(const float4*)(src + c);
            unsigned int lo = (__float_as_uint(f.x) >> 16) |
                              (__float_as_uint(f.y) & 0xffff0000u);
            unsigned int hi = (__float_as_uint(f.z) >> 16) |
                              (__float_as_uint(f.w) & 0xffff0000u);
            *(unsigned long long*)&iob[0][xrow][xseg + c] =
                (unsigned long long)lo | ((unsigned long long)hi << 32);
        }
    }
    __syncthreads();

    // ======================= ENCODER =======================
    for (int t = 0; t < S2S_T; ++t) {
        const int cur = t & 1, nxt = cur ^ 1;

        // ---- issue next step's x loads early (latency hidden by WMMAs) ----
        float4 xstage[8];
        if (t + 1 < S2S_T) {
            const float* src = x + ((size_t)(n0 + xrow) * S2S_T + (t + 1)) * S2S_D + xseg;
            if (t + 2 < S2S_T) __builtin_prefetch(src + S2S_D, 0, 0);
#pragma unroll
            for (int c = 0; c < 8; ++c)
                xstage[c] = *(const float4*)(src + c * 4);
        }

        // ---- pre = x_t @ Wxh + h @ Whh   (M=16, N=128/wave, K=256+512) ----
        v8f acc[8];
#pragma unroll
        for (int nt = 0; nt < 8; ++nt) acc[nt] = vzero;

        for (int kt = 0; kt < S2S_D / 32; ++kt) {            // x part, K=256
            v16bf a = load_a_frag(&iob[cur][nl][kt * 32 + half * 8]);
#pragma unroll
            for (int nt = 0; nt < 8; ++nt) {
                v16bf b = load_b_frag(eWxhP +
                    (size_t)(((wave * 8 + nt) * (S2S_D / 32) + kt) * 512 + lane * 16));
                acc[nt] = WMMA_BF16(a, b, acc[nt]);
            }
        }
        for (int kt = 0; kt < S2S_H / 32; ++kt) {            // h part, K=512
            v16bf a = load_a_frag(&hbuf[cur][nl][kt * 32 + half * 8]);
#pragma unroll
            for (int nt = 0; nt < 8; ++nt) {
                v16bf b = load_b_frag(eWhhP +
                    (size_t)(((wave * 8 + nt) * (S2S_H / 32) + kt) * 512 + lane * 16));
                acc[nt] = WMMA_BF16(a, b, acc[nt]);
            }
        }

        // ---- pack next x tile into the other buffer (disjoint: no barrier) ----
        if (t + 1 < S2S_T) {
#pragma unroll
            for (int c = 0; c < 8; ++c) {
                float4 f = xstage[c];
                unsigned int lo = (__float_as_uint(f.x) >> 16) |
                                  (__float_as_uint(f.y) & 0xffff0000u);
                unsigned int hi = (__float_as_uint(f.z) >> 16) |
                                  (__float_as_uint(f.w) & 0xffff0000u);
                *(unsigned long long*)&iob[nxt][xrow][xseg + c * 4] =
                    (unsigned long long)lo | ((unsigned long long)hi << 32);
            }
        }

        // ---- h_next = sigmoid(pre) -> hbuf[nxt] (disjoint buffer) ----
#pragma unroll
        for (int nt = 0; nt < 8; ++nt) {
            int col = wave * 128 + nt * 16 + nl;
#pragma unroll
            for (int r = 0; r < 8; ++r) {
                float sg = __builtin_amdgcn_rcpf(1.0f + __expf(-acc[nt][r]));
                hbuf[nxt][half * 8 + r][col] = f2bf_tr(sg);
            }
        }
        __syncthreads();   // hbuf[nxt] / iob[nxt] visible to all waves
    }

    // ======================= DECODER =======================
    // T even => decoder parity continues: step 0 reads hbuf[0] (encoder final h).
    // y_init = broadcast(y0) into iob[0].
    for (int i = tid; i < 16 * S2S_O; i += 128) {
        int row = i >> 8, col = i & 255;
        iob[0][row][col] = f2bf_tr(y0[col]);
    }
    __syncthreads();

    for (int t = 0; t < S2S_T; ++t) {
        const int cur = t & 1, nxt = cur ^ 1;

        // ---- s_next = sigmoid(y @ dWxh + s @ dWhh) ----
        v8f acc[8];
#pragma unroll
        for (int nt = 0; nt < 8; ++nt) acc[nt] = vzero;

        for (int kt = 0; kt < S2S_D / 32; ++kt) {
            v16bf a = load_a_frag(&iob[cur][nl][kt * 32 + half * 8]);
#pragma unroll
            for (int nt = 0; nt < 8; ++nt) {
                v16bf b = load_b_frag(dWxhP +
                    (size_t)(((wave * 8 + nt) * (S2S_D / 32) + kt) * 512 + lane * 16));
                acc[nt] = WMMA_BF16(a, b, acc[nt]);
            }
        }
        for (int kt = 0; kt < S2S_H / 32; ++kt) {
            v16bf a = load_a_frag(&hbuf[cur][nl][kt * 32 + half * 8]);
#pragma unroll
            for (int nt = 0; nt < 8; ++nt) {
                v16bf b = load_b_frag(dWhhP +
                    (size_t)(((wave * 8 + nt) * (S2S_H / 32) + kt) * 512 + lane * 16));
                acc[nt] = WMMA_BF16(a, b, acc[nt]);
            }
        }

#pragma unroll
        for (int nt = 0; nt < 8; ++nt) {
            int col = wave * 128 + nt * 16 + nl;
#pragma unroll
            for (int r = 0; r < 8; ++r) {
                float sg = __builtin_amdgcn_rcpf(1.0f + __expf(-acc[nt][r]));
                hbuf[nxt][half * 8 + r][col] = f2bf_tr(sg);
            }
        }
        __syncthreads();   // s ready in hbuf[nxt]

        // ---- y = s @ Who   (M=16, N=64/wave, K=512) ----
        v8f yacc[4];
#pragma unroll
        for (int nt = 0; nt < 4; ++nt) yacc[nt] = vzero;

        for (int kt = 0; kt < S2S_H / 32; ++kt) {
            v16bf a = load_a_frag(&hbuf[nxt][nl][kt * 32 + half * 8]);
#pragma unroll
            for (int nt = 0; nt < 4; ++nt) {
                v16bf b = load_b_frag(dWhoP +
                    (size_t)(((wave * 4 + nt) * (S2S_H / 32) + kt) * 512 + lane * 16));
                yacc[nt] = WMMA_BF16(a, b, yacc[nt]);
            }
        }

        // ---- write y_hat[t] (fp32) and feed y back into iob[nxt] ----
#pragma unroll
        for (int nt = 0; nt < 4; ++nt) {
            int oc = wave * 64 + nt * 16 + nl;
#pragma unroll
            for (int r = 0; r < 8; ++r) {
                int   m = half * 8 + r;
                float v = yacc[nt][r];
                out[(size_t)t * S2S_N * S2S_O + (size_t)(n0 + m) * S2S_O + oc] = v;
                iob[nxt][m][oc] = f2bf_tr(v);
            }
        }
        __syncthreads();   // iob[nxt] ready for next step
    }
}

// ---------------------------------------------------------------------------
// Host-side launcher
// ---------------------------------------------------------------------------
extern "C" void kernel_launch(void* const* d_in, const int* in_sizes, int n_in,
                              void* d_out, int out_size, void* d_ws, size_t ws_size,
                              hipStream_t stream) {
    const float* x    = (const float*)d_in[0];   // (N,T,D)
    const float* y0   = (const float*)d_in[1];   // (1,D)
    const float* eWxh = (const float*)d_in[2];   // (D,H)
    const float* eWhh = (const float*)d_in[3];   // (H,H)
    const float* dWxh = (const float*)d_in[4];   // (D,H)
    const float* dWhh = (const float*)d_in[5];   // (H,H)
    const float* dWho = (const float*)d_in[6];   // (H,O)

    // Workspace: packed bf16 weights (~1.75 MB total)
    unsigned short* ws    = (unsigned short*)d_ws;
    unsigned short* eWxhP = ws;
    unsigned short* eWhhP = eWxhP + S2S_D * S2S_H;
    unsigned short* dWxhP = eWhhP + S2S_H * S2S_H;
    unsigned short* dWhhP = dWxhP + S2S_D * S2S_H;
    unsigned short* dWhoP = dWhhP + S2S_H * S2S_H;

    pack_b_kernel<<<(S2S_D * S2S_H + 255) / 256, 256, 0, stream>>>(eWxh, eWxhP, S2S_D, S2S_H);
    pack_b_kernel<<<(S2S_H * S2S_H + 255) / 256, 256, 0, stream>>>(eWhh, eWhhP, S2S_H, S2S_H);
    pack_b_kernel<<<(S2S_D * S2S_H + 255) / 256, 256, 0, stream>>>(dWxh, dWxhP, S2S_D, S2S_H);
    pack_b_kernel<<<(S2S_H * S2S_H + 255) / 256, 256, 0, stream>>>(dWhh, dWhhP, S2S_H, S2S_H);
    pack_b_kernel<<<(S2S_H * S2S_O + 255) / 256, 256, 0, stream>>>(dWho, dWhoP, S2S_H, S2S_O);

    seq2seq_main_kernel<<<S2S_N / 16, 128, 0, stream>>>(
        x, y0, eWxhP, eWhhP, dWxhP, dWhhP, dWhoP, (float*)d_out);
}
